// DecoderBlock_4063039062783
// MI455X (gfx1250) — matrix operand order repaired and checked by reading, compiled, not run on hardware
//
#include <hip/hip_runtime.h>
#include <hip/hip_bf16.h>
#include <cmath>

#define NRES 4096
#define NATM 5
#define LDIM 256
#define PDIM 64
#define HN   8
#define DH   32
#define KNB  64
#define SIGMA_SCALE 10.0f

typedef _Float16 f16;
typedef __attribute__((ext_vector_type(16))) _Float16 v16h;
typedef __attribute__((ext_vector_type(8)))  float    v8f;

// ---------------------------------------------------------------- helpers

__device__ inline float gelu_f(float x){
    // jax.nn.gelu tanh approximation; tanh(y) = 1 - 2/(exp(2y)+1) (branch-free)
    const float c0 = 0.7978845608028654f, c1 = 0.044715f;
    float y = c0 * (x + c1 * x * x * x);
    float th = 1.0f - 2.0f / (__expf(2.0f * y) + 1.0f);
    return 0.5f * x * (1.0f + th);
}

__device__ inline float rng_uniform(unsigned seed, unsigned a, unsigned b){
    unsigned x = a * 0x9E3779B1u ^ (b + 0x7F4A7C15u) * 0x85EBCA77u ^ seed;
    x ^= x >> 16; x *= 0x7FEB352Du;
    x ^= x >> 15; x *= 0x846CA68Bu;
    x ^= x >> 16;
    return ((float)(x >> 8) + 0.5f) * (1.0f / 16777216.0f);
}

// A fragment: 16x32 f16 tile, row-major (M x K) LDS with leading dim `ld`.
// lanes 0-15: M=0..15, K base 0 ; lanes 16-31: same M, K base +8.
// halves 0..7 -> K {0..7}, halves 8..15 -> K {16..23} (+lane-half offset).
// Per lane this is two contiguous 8-half (16B) runs -> 2x ds_load_b128.
__device__ inline v16h load_a_frag(const f16* base, int ld){
    const int lane  = threadIdx.x & 31;
    const int m     = lane & 15;
    const int khalf = (lane >> 4) << 3;   // 0 or 8
    v16h a;
    #pragma unroll
    for (int e = 0; e < 16; ++e){
        int g = e >> 1;
        int k = ((g < 4) ? (g * 2) : (16 + (g - 4) * 2)) + khalf + (e & 1);
        a[e] = base[m * ld + k];
    }
    return a;
}

// B fragment from a TRANSPOSED (N-major) tile: LDS holds [n][k] with leading
// dim `ldk`. lanes 0-15: N=lane, K={0..7,16..23} ; lanes 16-31: same N, K={8..15,24..31}.
// Two contiguous 8-half runs per lane -> vectorized ds loads.
__device__ inline v16h load_bt_frag(const f16* base, int ldk){
    const int lane = threadIdx.x & 31;
    const int n    = lane & 15;
    const int kb   = (lane >> 4) << 3;    // 0 or 8
    v16h b;
    #pragma unroll
    for (int e = 0; e < 16; ++e)
        b[e] = base[n * ldk + kb + ((e < 8) ? e : e + 8)];
    return b;
}

// block-wide min extraction over key[NRES] with stable (index) tie-break.
__device__ inline void block_min(const float* key, float* rv, int* ri, int t){
    float bv = __builtin_inff(); int bj = NRES;
    for (int s = 0; s < NRES / 256; ++s){
        int j = t + s * 256;
        float v = key[j];
        if (v < bv || (v == bv && j < bj)){ bv = v; bj = j; }
    }
    rv[t] = bv; ri[t] = bj; __syncthreads();
    for (int off = 128; off > 0; off >>= 1){
        if (t < off && (rv[t+off] < rv[t] || (rv[t+off] == rv[t] && ri[t+off] < ri[t]))){
            rv[t] = rv[t+off]; ri[t] = ri[t+off];
        }
        __syncthreads();
    }
}

// ---------------------------------------------------------------- frames

__global__ void frames_kernel(const float* __restrict__ pos, float* __restrict__ Rg,
                              float* __restrict__ cag, float* __restrict__ lpF,
                              f16* __restrict__ lpH)
{
    int i = blockIdx.x * blockDim.x + threadIdx.x;
    if (i >= NRES) return;
    float nA[3], caA[3], cA[3];
    #pragma unroll
    for (int d = 0; d < 3; ++d){
        nA[d]  = pos[(i*NATM + 0)*3 + d];
        caA[d] = pos[(i*NATM + 1)*3 + d];
        cA[d]  = pos[(i*NATM + 2)*3 + d];
    }
    float e1[3], e2v[3], e3v[3], tv[3], uv[3];
    #pragma unroll
    for (int d = 0; d < 3; ++d) tv[d] = cA[d] - caA[d];
    float n1 = sqrtf(tv[0]*tv[0] + tv[1]*tv[1] + tv[2]*tv[2]) + 1e-8f;
    #pragma unroll
    for (int d = 0; d < 3; ++d) e1[d] = tv[d] / n1;
    #pragma unroll
    for (int d = 0; d < 3; ++d) uv[d] = nA[d] - caA[d];
    float pr = uv[0]*e1[0] + uv[1]*e1[1] + uv[2]*e1[2];
    #pragma unroll
    for (int d = 0; d < 3; ++d) tv[d] = uv[d] - pr * e1[d];
    float n2 = sqrtf(tv[0]*tv[0] + tv[1]*tv[1] + tv[2]*tv[2]) + 1e-8f;
    #pragma unroll
    for (int d = 0; d < 3; ++d) e2v[d] = tv[d] / n2;
    e3v[0] = e1[1]*e2v[2] - e1[2]*e2v[1];
    e3v[1] = e1[2]*e2v[0] - e1[0]*e2v[2];
    e3v[2] = e1[0]*e2v[1] - e1[1]*e2v[0];
    #pragma unroll
    for (int d = 0; d < 3; ++d){
        Rg[i*9 + d*3 + 0] = e1[d];
        Rg[i*9 + d*3 + 1] = e2v[d];
        Rg[i*9 + d*3 + 2] = e3v[d];
        cag[i*3 + d] = caA[d];
    }
    for (int a = 0; a < NATM; ++a){
        float rl[3];
        #pragma unroll
        for (int d = 0; d < 3; ++d) rl[d] = pos[(i*NATM + a)*3 + d] - caA[d];
        #pragma unroll
        for (int j = 0; j < 3; ++j){
            float s = 0.f;
            #pragma unroll
            for (int k = 0; k < 3; ++k) s += Rg[i*9 + k*3 + j] * rl[k];
            lpF[i*15 + a*3 + j] = s;
            lpH[i*32 + a*3 + j] = (f16)s;
        }
    }
    for (int e = 15; e < 32; ++e) lpH[i*32 + e] = (f16)0.f;
}

// ---------------------------------------------------------------- neighbours

__global__ void __launch_bounds__(256)
neighbours_cur_kernel(const float* __restrict__ cag, const int* __restrict__ resi,
                      const int* __restrict__ chain, const int* __restrict__ batch,
                      const unsigned char* __restrict__ mask, int* __restrict__ nbOut)
{
    const int i = blockIdx.x, t = threadIdx.x;
    __shared__ float key[NRES];
    __shared__ float rv[256];
    __shared__ int   ri[256];
    __shared__ int   sEx[16];
    __shared__ float sExV[16];

    const float cax = cag[i*3+0], cay = cag[i*3+1], caz = cag[i*3+2];
    const int ri0 = resi[i], ci0 = chain[i], bi0 = batch[i];
    const bool mi = mask[i] != 0;

    for (int s = 0; s < NRES/256; ++s){
        int j = t + s*256;
        bool sb = (batch[j] == bi0);
        bool sc = (chain[j] == ci0);
        bool valid = sb && mi && (mask[j] != 0);
        int dr = resi[j] - ri0; if (dr < 0) dr = -dr;
        bool within0 = (dr < 16) && sb && sc;
        float dx = cag[j*3+0]-cax, dy = cag[j*3+1]-cay, dz = cag[j*3+2]-caz;
        float d = sqrtf(dx*dx + dy*dy + dz*dz);
        key[j] = (!valid || within0) ? __builtin_inff() : d;
    }
    __syncthreads();

    // cutoff = 16th smallest of masked distance; mark 15 smallest as "within"
    for (int it = 0; it < 16; ++it){
        block_min(key, rv, ri, t);
        if (t == 0){
            sEx[it] = ri[0]; sExV[it] = rv[0];
            if (rv[0] < 3.0e38f) key[ri[0]] = __builtin_inff();
        }
        __syncthreads();
    }
    const float cutoff = sExV[15];
    if (t < 15 && sExV[t] < 3.0e38f) key[sEx[t]] = -1.0f;  // within sentinel
    __syncthreads();

    for (int s = 0; s < NRES/256; ++s){
        int j = t + s*256;
        bool sb = (batch[j] == bi0);
        bool sc = (chain[j] == ci0);
        bool valid = sb && mi && (mask[j] != 0);
        int dr = resi[j] - ri0; if (dr < 0) dr = -dr;
        bool within0 = (dr < 16) && sb && sc;
        float cur = key[j];
        float out;
        if (!valid) out = __builtin_inff();
        else if (within0 || cur == -1.0f || cur < cutoff) out = -10000.0f;
        else {
            float rd0 = -3.0f * __logf(fmaxf(cur, 1e-6f));
            float u = rng_uniform(0x1234567u, (unsigned)i, (unsigned)j);
            float g = -__logf(-__logf(u));
            out = g - rd0;              // = -(rd0 - gumbel)
        }
        key[j] = out;
    }
    __syncthreads();

    for (int it = 0; it < KNB; ++it){
        block_min(key, rv, ri, t);
        if (t == 0){
            float val = rv[0]; int idx = ri[0];
            nbOut[i*KNB + it] = (val < 3.0e38f && mi) ? idx : -1;
            key[idx < NRES ? idx : 0] = (idx < NRES) ? __builtin_inff() : key[0];
        }
        __syncthreads();
    }
}

__global__ void __launch_bounds__(256)
neighbours_dmap_kernel(const float* __restrict__ dmap, const int* __restrict__ batch,
                       const unsigned char* __restrict__ mask, int* __restrict__ nbOut)
{
    const int i = blockIdx.x, t = threadIdx.x;
    __shared__ float key[NRES];
    __shared__ float rv[256];
    __shared__ int   ri[256];
    const int bi0 = batch[i];
    const bool mi = mask[i] != 0;

    for (int s = 0; s < NRES/256; ++s){
        int j = t + s*256;
        bool valid = (batch[j] == bi0) && mi && (mask[j] != 0);
        float w0 = -3.0f * __logf(dmap[(size_t)i*NRES + j] + 1e-6f);
        float u  = rng_uniform(0x9E3779Bu, (unsigned)i, (unsigned)j);
        float d  = -w0 + __logf(-__logf(u));     // -(w0 + gumbel)
        key[j] = valid ? d : __builtin_inff();
    }
    __syncthreads();
    for (int it = 0; it < KNB; ++it){
        block_min(key, rv, ri, t);
        if (t == 0){
            float val = rv[0]; int idx = ri[0];
            nbOut[i*KNB + it] = (val < 3.0e38f && mi) ? idx : -1;
            key[idx < NRES ? idx : 0] = (idx < NRES) ? __builtin_inff() : key[0];
        }
        __syncthreads();
    }
}

// ------------------------------------------------ fused pair features -> head bias
// Per residue: build 64x64 edge features, LN, WMMA MLP 64->128->64 (gelu),
// project through wb (64x8) into per-head attention bias. Nothing big hits HBM.
// W1/W2 kept N-major (transposed) in LDS so B fragments are contiguous.

__global__ void __launch_bounds__(256)
pair_bias_kernel(const float* __restrict__ pos, const float* __restrict__ dmap,
                 const int* __restrict__ nb, const int* __restrict__ resi,
                 const int* __restrict__ chain, const unsigned char* __restrict__ mask,
                 const float* __restrict__ Rg, const float* __restrict__ cag,
                 const float* __restrict__ w_relpos, const float* __restrict__ w_rbf,
                 const float* __restrict__ w_dist, const float* __restrict__ w_dir,
                 const float* __restrict__ w_rot, const float* __restrict__ w_pv,
                 const float* __restrict__ ln_s, const float* __restrict__ ln_b,
                 const float* __restrict__ w1, const float* __restrict__ b1,
                 const float* __restrict__ w2, const float* __restrict__ b2,
                 const float* __restrict__ wb,
                 float* __restrict__ biasOut, int* __restrict__ pmOut)
{
    const int i = blockIdx.x;
    const int t = threadIdx.x;
    __shared__ f16 W1s[128*64];   // transposed: [n=0..127][k=0..63]
    __shared__ f16 W2s[64*128];   // transposed: [n=0..63][k=0..127]
    __shared__ f16 As[64*64];     // row-major (edge x P)
    __shared__ __align__(16) char ubuf[64*128*2];  // feat f32 (64x64) then Hs f16 (64x128)
    float* feat = (float*)ubuf;
    f16*   Hs   = (f16*)ubuf;
    __shared__ float sB1[128], sB2[64], sLnS[64], sLnB[64], sWb[512], sBias[512];
    __shared__ float sRi[9], sCai[3], sMean[64], sRstd[64];

    for (int idx = t; idx < 8192; idx += 256)           // W1s[n*64+k] = w1[k*128+n]
        W1s[idx] = (f16)w1[(idx & 63) * 128 + (idx >> 6)];
    for (int idx = t; idx < 8192; idx += 256)           // W2s[n*128+k] = w2[k*64+n]
        W2s[idx] = (f16)w2[(idx & 127) * 64 + (idx >> 7)];
    for (int idx = t; idx < 512;  idx += 256){ sWb[idx] = wb[idx]; sBias[idx] = 0.f; }
    if (t < 128) sB1[t] = b1[t];
    if (t < 64){ sB2[t] = b2[t]; sLnS[t] = ln_s[t]; sLnB[t] = ln_b[t]; }
    if (t < 9)  sRi[t]  = Rg[i*9 + t];
    if (t < 3)  sCai[t] = cag[i*3 + t];
    __syncthreads();

    { // raw edge features: thread t -> edge m = t/4, dims [g*16, g*16+16)
        const int m = t >> 2, g = t & 3;
        const int nv = nb[i*64 + m];
        const int jc = nv < 0 ? 0 : nv;
        const bool mi = mask[i] != 0;
        const bool pmf = mi && (mask[jc] != 0) && (nv != -1);
        if (g == 0) pmOut[i*64 + m] = pmf ? 1 : 0;
        int rel = resi[jc] - resi[i];
        rel = rel < -32 ? -32 : (rel > 32 ? 32 : rel);
        rel += 32;
        const bool sch = (chain[jc] == chain[i]) && pmf;

        float dv[3], Rj[9], dirn[3], rot[9], pv[15];
        #pragma unroll
        for (int d = 0; d < 3; ++d) dv[d] = cag[jc*3 + d] - sCai[d];
        const float dca = sqrtf(dv[0]*dv[0] + dv[1]*dv[1] + dv[2]*dv[2]);
        #pragma unroll
        for (int d = 0; d < 9; ++d) Rj[d] = Rg[jc*9 + d];
        float dn = 1e-8f;
        #pragma unroll
        for (int a = 0; a < 3; ++a){
            float s = 0.f;
            #pragma unroll
            for (int k = 0; k < 3; ++k) s += sRi[k*3 + a] * dv[k];
            dirn[a] = s;
        }
        dn += sqrtf(dirn[0]*dirn[0] + dirn[1]*dirn[1] + dirn[2]*dirn[2]);
        #pragma unroll
        for (int a = 0; a < 3; ++a) dirn[a] /= dn;
        #pragma unroll
        for (int a = 0; a < 3; ++a)
            #pragma unroll
            for (int b = 0; b < 3; ++b){
                float s = 0.f;
                #pragma unroll
                for (int k = 0; k < 3; ++k) s += sRi[k*3 + a] * Rj[k*3 + b];
                rot[a*3 + b] = s;
            }
        #pragma unroll
        for (int a = 0; a < NATM; ++a){
            float rl[3];
            #pragma unroll
            for (int d = 0; d < 3; ++d) rl[d] = pos[(jc*NATM + a)*3 + d] - sCai[d];
            #pragma unroll
            for (int jx = 0; jx < 3; ++jx){
                float s = 0.f;
                #pragma unroll
                for (int k = 0; k < 3; ++k) s += sRi[k*3 + jx] * rl[k];
                pv[a*3 + jx] = s;
            }
        }
        const float dm = dmap[(size_t)i*NRES + jc];
        float rb_dm[16], rb_dc[16];
        #pragma unroll
        for (int r = 0; r < 16; ++r){
            float mu = 22.0f * (float)r / 15.0f;
            float a1 = (dm  - mu) * (16.0f / 22.0f);
            float a2 = (dca - mu) * (16.0f / 22.0f);
            rb_dm[r] = __expf(-a1*a1);
            rb_dc[r] = __expf(-a2*a2);
        }
        for (int pp = g*16; pp < g*16 + 16; ++pp){
            float acc = sch ? w_relpos[rel*64 + pp] : 0.f;
            if (pmf){
                float s = 0.f;
                #pragma unroll
                for (int r = 0; r < 16; ++r) s += rb_dm[r] * w_rbf[r*64 + pp];
                acc += s;
            }
            float s2 = 0.f;
            #pragma unroll
            for (int r = 0; r < 16; ++r) s2 += rb_dc[r] * w_dist[r*64 + pp];
            acc += s2;
            #pragma unroll
            for (int r = 0; r < 3; ++r)  acc += dirn[r] * w_dir[r*64 + pp];
            #pragma unroll
            for (int r = 0; r < 9; ++r)  acc += rot[r]  * w_rot[r*64 + pp];
            #pragma unroll
            for (int r = 0; r < 15; ++r) acc += pv[r]   * w_pv[r*64 + pp];
            feat[m*64 + pp] = acc;
        }
    }
    __syncthreads();

    if (t < 64){ // LayerNorm stats per edge
        float s = 0.f, s2 = 0.f;
        for (int pp = 0; pp < 64; ++pp){ float v = feat[t*64 + pp]; s += v; s2 += v*v; }
        float mean = s * (1.0f/64.0f);
        float var  = s2 * (1.0f/64.0f) - mean*mean;
        sMean[t] = mean; sRstd[t] = rsqrtf(var + 1e-5f);
    }
    __syncthreads();
    for (int idx = t; idx < 4096; idx += 256){
        int m = idx >> 6, pp = idx & 63;
        As[idx] = (f16)((feat[idx] - sMean[m]) * sRstd[m] * sLnS[pp] + sLnB[pp]);
    }
    __syncthreads();

    const int wave = t >> 5, lane = t & 31;
    const int rowoff = (lane >= 16) ? 8 : 0;
    const int ncol = lane & 15;

    // GEMM1: Hs(64x128) = gelu(As @ W1 + b1), WMMA f16 -> f32 acc
    #pragma unroll
    for (int qq = 0; qq < 4; ++qq){
        int tile = wave*4 + qq;
        int tm = tile >> 3, tn = tile & 7;
        v8f acc = {};
        #pragma unroll
        for (int ks = 0; ks < 2; ++ks){
            v16h af = load_a_frag(&As[(tm*16)*64 + ks*32], 64);
            v16h bf = load_bt_frag(&W1s[(tn*16)*64 + ks*32], 64);
            acc = __builtin_amdgcn_wmma_f32_16x16x32_f16(false, af, false, bf,
                                                         (short)0, acc, false, false);
        }
        #pragma unroll
        for (int r = 0; r < 8; ++r){
            int mrow = tm*16 + rowoff + r;
            int nc   = tn*16 + ncol;
            Hs[mrow*128 + nc] = (f16)gelu_f(acc[r] + sB1[nc]);
        }
    }
    __syncthreads();

    // GEMM2: out(64x64) = Hs @ W2 + b2, folded straight into wb (64x8) head bias
    #pragma unroll
    for (int qq = 0; qq < 2; ++qq){
        int tile = wave*2 + qq;
        int tm = tile >> 2, tn = tile & 3;
        v8f acc = {};
        #pragma unroll
        for (int ks = 0; ks < 4; ++ks){
            v16h af = load_a_frag(&Hs[(tm*16)*128 + ks*32], 128);
            v16h bf = load_bt_frag(&W2s[(tn*16)*128 + ks*32], 128);
            acc = __builtin_amdgcn_wmma_f32_16x16x32_f16(false, af, false, bf,
                                                         (short)0, acc, false, false);
        }
        #pragma unroll
        for (int r = 0; r < 8; ++r){
            int mrow = tm*16 + rowoff + r;
            int pc   = tn*16 + ncol;
            float val = acc[r] + sB2[pc];
            #pragma unroll
            for (int hh = 0; hh < 8; ++hh)
                atomicAdd(&sBias[mrow*8 + hh], val * sWb[pc*8 + hh]);
        }
    }
    __syncthreads();
    for (int idx = t; idx < 512; idx += 256)
        biasOut[(size_t)i*512 + idx] = sBias[idx];
}

// ---------------------------------------------------------------- generic WMMA GEMM
// C(MxNc) = [gelu]( A(MxK,f16) @ B + bias ) + residual, B given TRANSPOSED
// (Nc x K row-major, i.e. column-major B) so both LDS tiles vectorize.

__global__ void __launch_bounds__(256)
gemm_wmma(const f16* __restrict__ A, const f16* __restrict__ BT,
          float* __restrict__ outF, f16* __restrict__ outH,
          const float* __restrict__ bias, const float* __restrict__ residual,
          int M, int K, int Nc, int act)
{
    __shared__ f16 As[64*32];     // [m][k]
    __shared__ f16 Bs[64*32];     // [n][k]  (transposed tile)
    const int m0 = blockIdx.x * 64;
    const int n0 = blockIdx.y * 64;
    const int t = threadIdx.x;
    const int wave = t >> 5, lane = t & 31;
    const int tr = wave >> 1;     // 0..3: 16-row slab
    const int tcp = wave & 1;     // 0..1: 32-col slab
    v8f acc0 = {}; v8f acc1 = {};

    for (int k0 = 0; k0 < K; k0 += 32){
        if (k0 + 32 < K) __builtin_prefetch(&A[(size_t)m0*K + k0 + 32], 0, 1);
        #pragma unroll
        for (int r = 0; r < 8; ++r){
            int idx = t*8 + r;
            int row = idx >> 5, col = idx & 31;
            As[idx] = A[(size_t)(m0 + row)*K + (k0 + col)];
        }
        #pragma unroll
        for (int r = 0; r < 8; ++r){
            int idx = t*8 + r;
            int n = idx >> 5, k = idx & 31;
            int gn = n0 + n;
            Bs[idx] = (gn < Nc) ? BT[(size_t)gn*K + (k0 + k)] : (f16)0.f;
        }
        __syncthreads();
        v16h af  = load_a_frag(&As[(tr*16)*32], 32);
        v16h bf0 = load_bt_frag(&Bs[(tcp*32)*32], 32);
        v16h bf1 = load_bt_frag(&Bs[(tcp*32 + 16)*32], 32);
        acc0 = __builtin_amdgcn_wmma_f32_16x16x32_f16(false, af, false, bf0, (short)0, acc0, false, false);
        acc1 = __builtin_amdgcn_wmma_f32_16x16x32_f16(false, af, false, bf1, (short)0, acc1, false, false);
        __syncthreads();
    }
    const int rowoff = (lane >= 16) ? 8 : 0;
    const int ncol = lane & 15;
    #pragma unroll
    for (int tt = 0; tt < 2; ++tt){
        int n = n0 + tcp*32 + tt*16 + ncol;
        if (n >= Nc) continue;
        #pragma unroll
        for (int r = 0; r < 8; ++r){
            float val = tt ? acc1[r] : acc0[r];
            int m = m0 + tr*16 + rowoff + r;
            if (bias) val += bias[n];
            if (act == 1) val = gelu_f(val);
            if (residual) val += residual[(size_t)m*Nc + n];
            if (outF) outF[(size_t)m*Nc + n] = val;
            if (outH) outH[(size_t)m*Nc + n] = (f16)val;
        }
    }
}

// ---------------------------------------------------------------- layernorm (rows of 256)

__global__ void __launch_bounds__(256)
layernorm_kernel(const float* __restrict__ x, const float* __restrict__ s,
                 const float* __restrict__ b, float* __restrict__ outF,
                 f16* __restrict__ outH)
{
    const int i = blockIdx.x, t = threadIdx.x;
    __shared__ float red[256];
    __shared__ float sm, sr;
    float v = x[(size_t)i*LDIM + t];
    red[t] = v; __syncthreads();
    for (int off = 128; off > 0; off >>= 1){
        if (t < off) red[t] += red[t + off];
        __syncthreads();
    }
    if (t == 0) sm = red[0] * (1.0f / LDIM);
    __syncthreads();
    float d = v - sm;
    red[t] = d*d; __syncthreads();
    for (int off = 128; off > 0; off >>= 1){
        if (t < off) red[t] += red[t + off];
        __syncthreads();
    }
    if (t == 0) sr = rsqrtf(red[0] * (1.0f / LDIM) + 1e-5f);
    __syncthreads();
    float y = d * sr * s[t] + b[t];
    if (outF) outF[(size_t)i*LDIM + t] = y;
    if (outH) outH[(size_t)i*LDIM + t] = (f16)y;
}

// ---------------------------------------------------------------- sparse attention core

__global__ void __launch_bounds__(256)
sparse_attn_kernel(const float* __restrict__ q, const float* __restrict__ k,
                   const float* __restrict__ v, const int* __restrict__ nb,
                   const int* __restrict__ pm, const float* __restrict__ biasP,
                   f16* __restrict__ aoH)
{
    const int i = blockIdx.x, t = threadIdx.x;
    const int h = t >> 5, lane = t & 31;
    __shared__ int   sJ[64];
    __shared__ int   sPM[64];
    __shared__ float sQ[256];
    __shared__ float sP[8][64];
    if (t < 64){
        int nv = nb[i*64 + t];
        sJ[t] = nv < 0 ? 0 : nv;
        sPM[t] = pm[i*64 + t];
    }
    sQ[t] = q[(size_t)i*256 + t];
    __syncthreads();

    float l0, l1;
    {
        int kk = lane, j = sJ[kk];
        float acc = 0.f;
        for (int d = 0; d < 32; ++d) acc += sQ[h*32 + d] * k[(size_t)j*256 + h*32 + d];
        l0 = sPM[kk] ? (acc * 0.17677669529663687f + biasP[(size_t)i*512 + kk*8 + h]) : -1e9f;
    }
    {
        int kk = lane + 32, j = sJ[kk];
        float acc = 0.f;
        for (int d = 0; d < 32; ++d) acc += sQ[h*32 + d] * k[(size_t)j*256 + h*32 + d];
        l1 = sPM[kk] ? (acc * 0.17677669529663687f + biasP[(size_t)i*512 + kk*8 + h]) : -1e9f;
    }
    float mx = fmaxf(l0, l1);
    for (int off = 16; off > 0; off >>= 1) mx = fmaxf(mx, __shfl_xor(mx, off, 32));
    float e0 = __expf(l0 - mx), e1 = __expf(l1 - mx);
    float sum = e0 + e1;
    for (int off = 16; off > 0; off >>= 1) sum += __shfl_xor(sum, off, 32);
    float inv = 1.0f / sum;
    sP[h][lane] = e0 * inv; sP[h][lane + 32] = e1 * inv;
    __syncthreads();
    float o = 0.f;
    for (int kk = 0; kk < 64; ++kk)
        o += sP[h][kk] * v[(size_t)sJ[kk]*256 + h*32 + lane];
    aoH[(size_t)i*256 + h*32 + lane] = (f16)o;
}

// ---------------------------------------------------------------- decoder helpers

__global__ void zero_kernel(float* __restrict__ p, int n){
    int i = blockIdx.x * blockDim.x + threadIdx.x;
    if (i < n) p[i] = 0.f;
}

// transpose + pad + f32->f16: dst is (dN x dK) row-major, dst[n*dK+k] = src[k*sN+n]
__global__ void convert_pad_t_kernel(const float* __restrict__ src, f16* __restrict__ dst,
                                     int sK, int sN, int dN, int dK)
{
    int idx = blockIdx.x * blockDim.x + threadIdx.x;
    if (idx >= dN * dK) return;
    int n = idx / dK, k = idx % dK;
    dst[idx] = (k < sK && n < sN) ? (f16)src[(size_t)k*sN + n] : (f16)0.f;
}

__global__ void segsum_scatter_kernel(const float* __restrict__ u, const float* __restrict__ bgg,
                                      const float* __restrict__ cgg, const int* __restrict__ batch,
                                      const int* __restrict__ chain, const unsigned char* __restrict__ mask,
                                      float* __restrict__ numB, float* __restrict__ numC)
{
    int idx = blockIdx.x * blockDim.x + threadIdx.x;
    if (idx >= NRES * 512) return;
    int i = idx >> 9, c = idx & 511;
    float m = mask[i] ? 1.f : 0.f;
    float uv = u[idx] * m;
    atomicAdd(&numB[batch[i]*512 + c], bgg[idx] * uv);
    atomicAdd(&numC[chain[i]*512 + c], cgg[idx] * uv);
}

__global__ void segden_kernel(const int* __restrict__ batch, const int* __restrict__ chain,
                              const unsigned char* __restrict__ mask,
                              float* __restrict__ denB, float* __restrict__ denC)
{
    int i = blockIdx.x * blockDim.x + threadIdx.x;
    if (i >= NRES) return;
    float m = mask[i] ? 1.f : 0.f;
    atomicAdd(&denB[batch[i]], m);
    atomicAdd(&denC[chain[i]], m);
}

__global__ void hidden_combine_kernel(const float* __restrict__ u, const float* __restrict__ lgg,
                                      const float* __restrict__ numB, const float* __restrict__ denB,
                                      const float* __restrict__ numC, const float* __restrict__ denC,
                                      const int* __restrict__ batch, const int* __restrict__ chain,
                                      f16* __restrict__ hiddenH)
{
    int idx = blockIdx.x * blockDim.x + threadIdx.x;
    if (idx >= NRES * 512) return;
    int i = idx >> 9, c = idx & 511;
    float hb = numB[batch[i]*512 + c] / fmaxf(denB[batch[i]], 1e-6f);
    float hc = numC[chain[i]*512 + c] / fmaxf(denC[chain[i]], 1e-6f);
    hiddenH[idx] = (f16)(hb + hc + lgg[idx] * u[idx]);
}

__global__ void update_pos_kernel(const float* __restrict__ Rg, const float* __restrict__ cag,
                                  const float* __restrict__ lpF, const float* __restrict__ updP,
                                  float* __restrict__ outPos)
{
    int i = blockIdx.x * blockDim.x + threadIdx.x;
    if (i >= NRES) return;
    float lp2[15];
    #pragma unroll
    for (int e = 0; e < 15; ++e) lp2[e] = lpF[i*15 + e] + SIGMA_SCALE * updP[i*16 + e];
    for (int a = 0; a < NATM; ++a){
        #pragma unroll
        for (int ii = 0; ii < 3; ++ii){
            float s = cag[i*3 + ii];
            #pragma unroll
            for (int j = 0; j < 3; ++j) s += Rg[i*9 + ii*3 + j] * lp2[a*3 + j];
            outPos[(size_t)(i*NATM + a)*3 + ii] = s;
        }
    }
}

// ---------------------------------------------------------------- launcher

extern "C" void kernel_launch(void* const* d_in, const int* in_sizes, int n_in,
                              void* d_out, int out_size, void* d_ws, size_t ws_size,
                              hipStream_t stream)
{
    (void)in_sizes; (void)n_in; (void)out_size; (void)ws_size;
    const float* features = (const float*)d_in[0];
    const float* pos      = (const float*)d_in[1];
    const float* dmap     = (const float*)d_in[2];
    const int*   resi     = (const int*)d_in[3];
    const int*   chain    = (const int*)d_in[4];
    const int*   batch    = (const int*)d_in[5];
    const unsigned char* mask = (const unsigned char*)d_in[6];
    #define PRM(i) ((const float*)d_in[(i)])
    // params leaves (insertion order): pf1 7..18, pf2 19..30, att1 31..35,
    // att2 36..40, upd 41..50, ln1..ln4 51..58, w_pos 59

    char* wp = (char*)d_ws;
    auto carve = [&](size_t bytes)->char*{
        char* r = wp; wp += (bytes + 255) & ~(size_t)255; return r;
    };
    float* Rg     = (float*)carve((size_t)NRES*9*4);
    float* cag    = (float*)carve((size_t)NRES*3*4);
    float* lpF    = (float*)carve((size_t)NRES*15*4);
    f16*   lpH    = (f16*)  carve((size_t)NRES*32*2);
    int*   nb1    = (int*)  carve((size_t)NRES*64*4);
    int*   nb2    = (int*)  carve((size_t)NRES*64*4);
    int*   pm1    = (int*)  carve((size_t)NRES*64*4);
    int*   pm2    = (int*)  carve((size_t)NRES*64*4);
    float* bias1  = (float*)carve((size_t)NRES*512*4);
    float* bias2  = (float*)carve((size_t)NRES*512*4);
    f16*   lnH    = (f16*)  carve((size_t)NRES*256*2);
    float* qF     = (float*)carve((size_t)NRES*256*4);
    float* kF     = (float*)carve((size_t)NRES*256*4);
    float* vF     = (float*)carve((size_t)NRES*256*4);
    f16*   aoH    = (f16*)  carve((size_t)NRES*256*2);
    float* x1     = (float*)carve((size_t)NRES*256*4);
    float* x2     = (float*)carve((size_t)NRES*256*4);
    float* local3 = (float*)carve((size_t)NRES*256*4);
    f16*   hid1H  = (f16*)  carve((size_t)NRES*512*2);
    f16*   local2H= (f16*)  carve((size_t)NRES*256*2);
    float* uF     = (float*)carve((size_t)NRES*512*4);
    float* lggF   = (float*)carve((size_t)NRES*512*4);
    float* cggF   = (float*)carve((size_t)NRES*512*4);
    float* bggF   = (float*)carve((size_t)NRES*512*4);
    float* segbuf = (float*)carve((size_t)(2*32*512 + 64)*4);
    float* numB = segbuf;
    float* numC = segbuf + 32*512;
    float* denB = segbuf + 2*32*512;
    float* denC = denB + 32;
    f16*   hiddenH= (f16*)  carve((size_t)NRES*512*2);
    float* updP   = (float*)carve((size_t)NRES*16*4);
    f16* wq1H = (f16*)carve(65536*2); f16* wk1H = (f16*)carve(65536*2);
    f16* wv1H = (f16*)carve(65536*2); f16* wo1H = (f16*)carve(65536*2);
    f16* wq2H = (f16*)carve(65536*2); f16* wk2H = (f16*)carve(65536*2);
    f16* wv2H = (f16*)carve(65536*2); f16* wo2H = (f16*)carve(65536*2);
    f16* wlp1H = (f16*)carve((size_t)512*32*2);
    f16* wlp2H = (f16*)carve((size_t)256*512*2);
    f16* wuH  = (f16*)carve((size_t)512*256*2);
    f16* wlgH = (f16*)carve((size_t)512*256*2);
    f16* wcgH = (f16*)carve((size_t)512*256*2);
    f16* wbgH = (f16*)carve((size_t)512*256*2);
    f16* woH  = (f16*)carve((size_t)256*512*2);
    f16* wposH= (f16*)carve((size_t)16*256*2);

    // transpose+pad+convert weights: dst (dN x dK) row-major = B^T
    auto convT = [&](const float* src, f16* dst, int sK, int sN, int dN, int dK){
        int n = dN * dK;
        convert_pad_t_kernel<<<(n + 255)/256, 256, 0, stream>>>(src, dst, sK, sN, dN, dK);
    };
    auto gemm = [&](const f16* A, const f16* BT, float* oF, f16* oH, const float* bias,
                    const float* res, int K, int Nc, int act){
        dim3 g(NRES/64, (Nc + 63)/64);
        gemm_wmma<<<g, 256, 0, stream>>>(A, BT, oF, oH, bias, res, NRES, K, Nc, act);
    };

    // geometry + weight precision conversion
    frames_kernel<<<NRES/256, 256, 0, stream>>>(pos, Rg, cag, lpF, lpH);
    convT(PRM(31), wq1H, 256,256,256,256); convT(PRM(32), wk1H, 256,256,256,256);
    convT(PRM(33), wv1H, 256,256,256,256); convT(PRM(35), wo1H, 256,256,256,256);
    convT(PRM(36), wq2H, 256,256,256,256); convT(PRM(37), wk2H, 256,256,256,256);
    convT(PRM(38), wv2H, 256,256,256,256); convT(PRM(40), wo2H, 256,256,256,256);
    convT(PRM(41), wlp1H, 15,512,512,32);
    convT(PRM(43), wlp2H, 512,256,256,512);
    convT(PRM(45), wuH, 256,512,512,256);  convT(PRM(46), wlgH, 256,512,512,256);
    convT(PRM(47), wcgH, 256,512,512,256); convT(PRM(48), wbgH, 256,512,512,256);
    convT(PRM(49), woH, 512,256,256,512);
    convT(PRM(59), wposH, 256,15,16,256);

    // neighbour selection (LDS-resident top-k, Gumbel perturbed)
    neighbours_cur_kernel<<<NRES, 256, 0, stream>>>(cag, resi, chain, batch, mask, nb1);
    neighbours_dmap_kernel<<<NRES, 256, 0, stream>>>(dmap, batch, mask, nb2);

    // fused pair features -> per-head attention bias
    pair_bias_kernel<<<NRES, 256, 0, stream>>>(pos, dmap, nb1, resi, chain, mask, Rg, cag,
        PRM(7), PRM(8), PRM(9), PRM(10), PRM(11), PRM(12), PRM(13), PRM(14),
        PRM(15), PRM(16), PRM(17), PRM(18), PRM(34), bias1, pm1);
    pair_bias_kernel<<<NRES, 256, 0, stream>>>(pos, dmap, nb2, resi, chain, mask, Rg, cag,
        PRM(19), PRM(20), PRM(21), PRM(22), PRM(23), PRM(24), PRM(25), PRM(26),
        PRM(27), PRM(28), PRM(29), PRM(30), PRM(39), bias2, pm2);

    // attention block 1
    layernorm_kernel<<<NRES, 256, 0, stream>>>(features, PRM(51), PRM(52), nullptr, lnH);
    gemm(lnH, wq1H, qF, nullptr, nullptr, nullptr, 256, 256, 0);
    gemm(lnH, wk1H, kF, nullptr, nullptr, nullptr, 256, 256, 0);
    gemm(lnH, wv1H, vF, nullptr, nullptr, nullptr, 256, 256, 0);
    sparse_attn_kernel<<<NRES, 256, 0, stream>>>(qF, kF, vF, nb1, pm1, bias1, aoH);
    gemm(aoH, wo1H, x1, nullptr, nullptr, features, 256, 256, 0);

    // attention block 2
    layernorm_kernel<<<NRES, 256, 0, stream>>>(x1, PRM(53), PRM(54), nullptr, lnH);
    gemm(lnH, wq2H, qF, nullptr, nullptr, nullptr, 256, 256, 0);
    gemm(lnH, wk2H, kF, nullptr, nullptr, nullptr, 256, 256, 0);
    gemm(lnH, wv2H, vF, nullptr, nullptr, nullptr, 256, 256, 0);
    sparse_attn_kernel<<<NRES, 256, 0, stream>>>(qF, kF, vF, nb2, pm2, bias2, aoH);
    gemm(aoH, wo2H, x2, nullptr, nullptr, x1, 256, 256, 0);

    // decoder update
    layernorm_kernel<<<NRES, 256, 0, stream>>>(x2, PRM(55), PRM(56), local3, nullptr);
    gemm(lpH, wlp1H, nullptr, hid1H, PRM(42), nullptr, 32, 512, 1);
    gemm(hid1H, wlp2H, nullptr, local2H, PRM(44), local3, 512, 256, 0);
    gemm(local2H, wuH,  uF,   nullptr, nullptr, nullptr, 256, 512, 0);
    gemm(local2H, wlgH, lggF, nullptr, nullptr, nullptr, 256, 512, 1);
    gemm(local2H, wcgH, cggF, nullptr, nullptr, nullptr, 256, 512, 1);
    gemm(local2H, wbgH, bggF, nullptr, nullptr, nullptr, 256, 512, 1);
    {
        int n = 2*32*512 + 64;
        zero_kernel<<<(n + 255)/256, 256, 0, stream>>>(segbuf, n);
    }
    segsum_scatter_kernel<<<(NRES*512 + 255)/256, 256, 0, stream>>>(uF, bggF, cggF, batch, chain, mask, numB, numC);
    segden_kernel<<<NRES/256, 256, 0, stream>>>(batch, chain, mask, denB, denC);
    hidden_combine_kernel<<<(NRES*512 + 255)/256, 256, 0, stream>>>(uF, lggF, numB, denB, numC, denC, batch, chain, hiddenH);
    gemm(hiddenH, woH, (float*)d_out, nullptr, PRM(50), x2, 512, 256, 0);  // final x

    // position update
    layernorm_kernel<<<NRES, 256, 0, stream>>>((const float*)d_out, PRM(57), PRM(58), nullptr, lnH);
    gemm(lnH, wposH, updP, nullptr, nullptr, nullptr, 256, 16, 0);
    update_pos_kernel<<<NRES/256, 256, 0, stream>>>(Rg, cag, lpF, updP, (float*)d_out + (size_t)NRES*LDIM);
    #undef PRM
}
